// IGKT_12326556139632
// MI455X (gfx1250) — matrix operand order repaired and checked by reading, not compiled, not run on hardware
//
#include <hip/hip_runtime.h>

typedef unsigned short u16;
typedef __attribute__((ext_vector_type(16))) __bf16 v16bf;
typedef __attribute__((ext_vector_type(8)))  float  v8f;

union V16BF { v16bf v; u16 u[16]; uint4 q[2]; };

__device__ __forceinline__ u16 f2bf(float f) {
    unsigned int x = __float_as_uint(f);
    unsigned int r = (x + 0x7FFFu + ((x >> 16) & 1u)) >> 16;   // RNE
    return (u16)r;
}
__device__ __forceinline__ float bf2f(u16 b) {
    return __uint_as_float(((unsigned int)b) << 16);
}

// ---------------------------------------------------------------- utilities
__global__ void cast_f32_bf16(const float* __restrict__ in, u16* __restrict__ out, int n) {
    int i = blockIdx.x * blockDim.x + threadIdx.x;
    if (i < n) out[i] = f2bf(in[i]);
}

__global__ void zero_f32(float* __restrict__ p, int n) {
    int i = blockIdx.x * blockDim.x + threadIdx.x;
    if (i < n) p[i] = 0.0f;
}

// Transposed basis combine: Wt[r][n][k] (bf16, [6,32,K]) so a WMMA B-fragment
// (column n, two runs of 8 consecutive K) is two contiguous 16-byte chunks.
// r<5: W[r]=sum_b wcomp[r,b]*bases[b];  r==5: loopw.
__global__ void combine_w_t(const float* __restrict__ bases, const float* __restrict__ wcomp,
                            const float* __restrict__ loopw, u16* __restrict__ Wt, int K) {
    int idx = blockIdx.x * blockDim.x + threadIdx.x;
    int tot = 6 * 32 * K;
    if (idx >= tot) return;
    int r = idx / (32 * K);
    int rem = idx - r * (32 * K);
    int n = rem / K;
    int k = rem - n * K;
    float v;
    if (r < 5) v = wcomp[r * 2 + 0] * bases[k * 32 + n] +
                   wcomp[r * 2 + 1] * bases[K * 32 + k * 32 + n];
    else       v = loopw[k * 32 + n];
    Wt[idx] = f2bf(v);
}

// lin1_w [256,128] f32 -> transposed bf16 [128,256]
__global__ void transpose_lin1(const float* __restrict__ w, u16* __restrict__ wt) {
    int idx = blockIdx.x * blockDim.x + threadIdx.x;   // 128*256
    if (idx >= 128 * 256) return;
    int n = idx >> 8, k = idx & 255;
    wt[idx] = f2bf(w[k * 128 + n]);
}

// -------------------------------------------------- R-GCN dense GEMM (WMMA)
// h_rel[r] = h * W[r]  for r=0..4 (bf16 out), r=5 -> loop buffer (f32 out).
// Wave per 16-row tile; two 16x16 n-tiles; KST fully-unrolled 16x16x32 WMMAs.
template <int KST>
__global__ void rgcn_gemm(const u16* __restrict__ hq,   // [N,K] bf16
                          const u16* __restrict__ Wt,   // [6,32,K] bf16 (n-major)
                          u16* __restrict__ hrel,       // [5,N,32] bf16
                          float* __restrict__ loopb,    // [N,32] f32
                          int Nn) {
    constexpr int K = KST * 32;
    const int r    = blockIdx.y;
    const int lane = threadIdx.x & 31;
    const int wave = threadIdx.x >> 5;
    const int nWav = gridDim.x * (blockDim.x >> 5);
    const int hf   = lane >> 4;
    const int l16  = lane & 15;
    const u16* Wr  = Wt + r * 32 * K;

    // B fragments fully in registers: column n = nt*16+l16,
    // half 0: K kk*32+{0..7,16..23}; half 1: +8 each (matches A layout).
    V16BF Bv[2][KST];
#pragma unroll
    for (int nt = 0; nt < 2; ++nt) {
        const u16* bcol = Wr + (nt * 16 + l16) * K;
#pragma unroll
        for (int kk = 0; kk < KST; ++kk) {
            const u16* bp = bcol + kk * 32 + hf * 8;
            Bv[nt][kk].q[0] = *(const uint4*)(bp);
            Bv[nt][kk].q[1] = *(const uint4*)(bp + 16);
        }
    }

    const int mtiles = Nn >> 4;
    for (int wid = blockIdx.x * (blockDim.x >> 5) + wave; wid < mtiles; wid += nWav) {
        const u16* Arow = hq + (size_t)(wid * 16 + l16) * K;
        v8f c0 = {}, c1 = {};
#pragma unroll
        for (int kk = 0; kk < KST; ++kk) {
            V16BF Av;
            // half 0: K kk*32+{0..7,16..23}; half 1: +8 each (ISA A layout)
            Av.q[0] = *(const uint4*)(Arow + kk * 32 + hf * 8);
            Av.q[1] = *(const uint4*)(Arow + kk * 32 + 16 + hf * 8);
            c0 = __builtin_amdgcn_wmma_f32_16x16x32_bf16(false, Av.v, false, Bv[0][kk].v,
                                                         (short)0, c0, false, false);
            c1 = __builtin_amdgcn_wmma_f32_16x16x32_bf16(false, Av.v, false, Bv[1][kk].v,
                                                         (short)0, c1, false, false);
        }
        // D layout: VGPR j -> row j + hf*8, col l16 (c0: n 0..15, c1: n 16..31)
        if (r < 5) {
            u16* outp = hrel + (size_t)r * Nn * 32;
#pragma unroll
            for (int j = 0; j < 8; ++j) {
                int m = wid * 16 + j + hf * 8;
                outp[(size_t)m * 32 + l16]      = f2bf(c0[j]);
                outp[(size_t)m * 32 + 16 + l16] = f2bf(c1[j]);
            }
        } else {
#pragma unroll
            for (int j = 0; j < 8; ++j) {
                int m = wid * 16 + j + hf * 8;
                loopb[(size_t)m * 32 + l16]      = c0[j];
                loopb[(size_t)m * 32 + 16 + l16] = c1[j];
            }
        }
    }
}

// -------------------------------------------------------------- edge phase
// 4 threads per edge; each gathers 8 bf16 feats (16B) and atomically adds.
__global__ void edge_scatter(const int* __restrict__ src, const int* __restrict__ dst,
                             const int* __restrict__ et, const float* __restrict__ mask,
                             const u16* __restrict__ hrel, float* __restrict__ agg,
                             int E, int Nn) {
    long long idx   = (long long)blockIdx.x * blockDim.x + threadIdx.x;
    long long total = (long long)E * 4;
    long long step  = (long long)gridDim.x * blockDim.x;
    for (; idx < total; idx += step) {
        int e = (int)(idx >> 2);
        int q = (int)(idx & 3);
        float m = mask[e];
        if (m == 0.0f) continue;
        int r = et[e], s = src[e], d = dst[e];
        uint4 pk = *(const uint4*)(hrel + ((size_t)r * Nn + s) * 32 + q * 8);
        const u16* u = (const u16*)&pk;
        float* ap = agg + (size_t)d * 32 + q * 8;
#pragma unroll
        for (int t = 0; t < 8; ++t) atomicAdd(ap + t, bf2f(u[t]) * m);
    }
}

// h_out = tanh(agg + loop + bias) -> concat buffer cs[:,col0:col0+32] and bf16 next input
__global__ void finalize_layer(const float* __restrict__ agg, const float* __restrict__ loopb,
                               const float* __restrict__ bias, float* __restrict__ cs, int col0,
                               u16* __restrict__ hq_next, int Nn) {
    int idx = blockIdx.x * blockDim.x + threadIdx.x;
    if (idx >= Nn * 32) return;
    int n = idx >> 5, j = idx & 31;
    float v = tanhf(agg[idx] + loopb[idx] + bias[j]);
    cs[(size_t)n * 128 + col0 + j] = v;
    hq_next[(size_t)n * 32 + j]    = f2bf(v);
}

// ----------------------------------------------------------------- MLP part
__global__ void gather_z(const float* __restrict__ cs, const int* __restrict__ ui,
                         const int* __restrict__ ii, u16* __restrict__ zq, int P) {
    int idx = blockIdx.x * blockDim.x + threadIdx.x;
    if (idx >= P * 256) return;
    int p = idx >> 8, j = idx & 255;
    int node = (j < 128) ? ui[p] : ii[p];
    zq[idx] = f2bf(cs[(size_t)node * 128 + (j & 127)]);
}

// hidden = relu(zq * lin1_w + b) : [M,256]x[256,128] bf16 WMMA, B transposed [128,256]
__global__ void mlp_gemm(const u16* __restrict__ Aq,   // [M,256] bf16
                         const u16* __restrict__ Bt,   // [128,256] bf16 (n-major)
                         const float* __restrict__ bias, float* __restrict__ Dm, int M) {
    const int lane = threadIdx.x & 31, hf = lane >> 4, l16 = lane & 15;
    const int wid  = blockIdx.x * (blockDim.x >> 5) + (threadIdx.x >> 5);
    const int nWav = gridDim.x * (blockDim.x >> 5);
    const int mtiles = M >> 4, ntiles = 8;
    for (int w = wid; w < mtiles * ntiles; w += nWav) {
        int mt = w / ntiles, nt = w - mt * ntiles;
        int n = nt * 16 + l16;
        const u16* Arow = Aq + (size_t)(mt * 16 + l16) * 256;
        const u16* Bcol = Bt + (size_t)n * 256;
        v8f c = {};
#pragma unroll
        for (int kk = 0; kk < 8; ++kk) {
            V16BF Av, Bv;
            Av.q[0] = *(const uint4*)(Arow + kk * 32 + hf * 8);
            Av.q[1] = *(const uint4*)(Arow + kk * 32 + 16 + hf * 8);
            const u16* bp = Bcol + kk * 32 + hf * 8;
            Bv.q[0] = *(const uint4*)(bp);
            Bv.q[1] = *(const uint4*)(bp + 16);
            c = __builtin_amdgcn_wmma_f32_16x16x32_bf16(false, Av.v, false, Bv.v,
                                                        (short)0, c, false, false);
        }
#pragma unroll
        for (int j = 0; j < 8; ++j) {
            int m = mt * 16 + j + hf * 8;
            float v = c[j] + bias[n];
            Dm[(size_t)m * 128 + n] = v > 0.0f ? v : 0.0f;
        }
    }
}

__global__ void final_out(const float* __restrict__ hidden, const float* __restrict__ w2,
                          const float* __restrict__ b2, float* __restrict__ outp, int P) {
    int p = blockIdx.x * blockDim.x + threadIdx.x;
    if (p >= P) return;
    const float* hp = hidden + (size_t)p * 128;
    float acc = 0.0f;
#pragma unroll 4
    for (int j = 0; j < 128; ++j) acc += hp[j] * w2[j];
    acc += b2[0];
    outp[p] = 1.0f / (1.0f + __expf(-acc));
}

// ------------------------------------------------------------------- driver
extern "C" void kernel_launch(void* const* d_in, const int* in_sizes, int n_in,
                              void* d_out, int out_size, void* d_ws, size_t ws_size,
                              hipStream_t stream) {
    const int Nn = in_sizes[0] / 64;   // 100000
    const int E  = in_sizes[1];        // 3.2M
    const int P  = in_sizes[5];        // 4096

    const float* x    = (const float*)d_in[0];
    const int* srcI   = (const int*)d_in[1];
    const int* dstI   = (const int*)d_in[2];
    const int* etI    = (const int*)d_in[3];
    const float* mask = (const float*)d_in[4];
    const int* uidx   = (const int*)d_in[5];
    const int* iidx   = (const int*)d_in[6];
    const float* lin1_w = (const float*)d_in[23];
    const float* lin1_b = (const float*)d_in[24];
    const float* lin2_w = (const float*)d_in[25];
    const float* lin2_b = (const float*)d_in[26];
    float* outp = (float*)d_out;

    char* wp = (char*)d_ws;
    auto carve = [&](size_t bytes) -> char* {
        char* r = wp; wp += (bytes + 255) & ~(size_t)255; return r;
    };
    float* cs    = (float*)carve((size_t)Nn * 128 * 4);   // concat states [N,128]
    float* agg   = (float*)carve((size_t)Nn * 32 * 4);
    float* loopb = (float*)carve((size_t)Nn * 32 * 4);
    u16*   hrel  = (u16*)  carve((size_t)5 * Nn * 32 * 2);
    u16*   hq    = (u16*)  carve((size_t)Nn * 64 * 2);    // layer input (bf16)
    u16*   Wt    = (u16*)  carve((size_t)6 * 32 * 64 * 2);
    u16*   zq    = (u16*)  carve((size_t)P * 256 * 2);
    u16*   w1t   = (u16*)  carve((size_t)128 * 256 * 2);
    float* hidden= (float*)carve((size_t)P * 128 * 4);

    cast_f32_bf16<<<(Nn * 64 + 255) / 256, 256, 0, stream>>>(x, hq, Nn * 64);

    for (int l = 0; l < 4; ++l) {
        const int K = (l == 0) ? 64 : 32;
        const float* bases = (const float*)d_in[7 + 4 * l];
        const float* wcomp = (const float*)d_in[8 + 4 * l];
        const float* loopw = (const float*)d_in[9 + 4 * l];
        const float* bias  = (const float*)d_in[10 + 4 * l];

        combine_w_t<<<(6 * 32 * K + 255) / 256, 256, 0, stream>>>(bases, wcomp, loopw, Wt, K);
        zero_f32<<<(Nn * 32 + 255) / 256, 256, 0, stream>>>(agg, Nn * 32);

        int mtiles  = Nn / 16;
        int blocksX = (mtiles + 7) / 8;   // 8 waves per 256-thread block
        if (l == 0)
            rgcn_gemm<2><<<dim3(blocksX, 6), 256, 0, stream>>>(hq, Wt, hrel, loopb, Nn);
        else
            rgcn_gemm<1><<<dim3(blocksX, 6), 256, 0, stream>>>(hq, Wt, hrel, loopb, Nn);

        edge_scatter<<<4096, 256, 0, stream>>>(srcI, dstI, etI, mask, hrel, agg, E, Nn);

        finalize_layer<<<(Nn * 32 + 255) / 256, 256, 0, stream>>>(agg, loopb, bias, cs,
                                                                  32 * l, hq, Nn);
    }

    transpose_lin1<<<(128 * 256 + 255) / 256, 256, 0, stream>>>(lin1_w, w1t);
    gather_z<<<(P * 256 + 255) / 256, 256, 0, stream>>>(cs, uidx, iidx, zq, P);
    int mlpWaves  = (P / 16) * 8;
    int mlpBlocks = (mlpWaves + 7) / 8;
    mlp_gemm<<<mlpBlocks, 256, 0, stream>>>(zq, w1t, lin1_b, hidden, P);
    final_out<<<(P + 255) / 256, 256, 0, stream>>>(hidden, lin2_w, lin2_b, outp, P);
}
